// DDSTSTransformer_9663676416718
// MI455X (gfx1250) — matrix operand order repaired and checked by reading, not compiled, not run on hardware
//
#include <hip/hip_runtime.h>
#include <math.h>
#include <stdint.h>

typedef __bf16 bf16_t;
typedef bf16_t bf16x16 __attribute__((ext_vector_type(16)));
typedef bf16_t bf16x8  __attribute__((ext_vector_type(8)));
typedef float  floatx8 __attribute__((ext_vector_type(8)));

constexpr int kB   = 32;
constexpr int kCIN = 6;
constexpr int kL   = 512;
constexpr int kTC  = 32;
constexpr int kH   = 8;
constexpr int kD   = 6;
constexpr int kHC  = kH * kTC;   // 256
constexpr int kFF  = 4 * kTC;    // 128

constexpr int QSTR = 40;         // sqkT row stride (bf16): 80 B, 16B-aligned rows
constexpr int SSTR = kL + 16;    // sv row stride: 1056 B
constexpr int PSTR = 40;         // P-tile row stride

// ISA 7.12.2 16-bit fragment packing (wave32):
//  A (16xK): lane half hi=((lane>>4)<<3); elems 0..7 -> K=hi..hi+7,
//            elems 8..15 -> K=16+hi..23+hi  => two contiguous 8-elem runs.
//  B (Kx16): same as A (koff>>1 == hi) => two contiguous 8-elem runs.

__device__ __forceinline__ void load_a_frag(bf16x16* f, const bf16_t* row, int hi) {
  ((bf16x8*)f)[0] = *(const bf16x8*)(row + hi);
  ((bf16x8*)f)[1] = *(const bf16x8*)(row + 16 + hi);
}
__device__ __forceinline__ void load_b_frag(bf16x16* f, const bf16_t* row, int koff) {
  ((bf16x8*)f)[0] = *(const bf16x8*)(row + (koff >> 1));
  ((bf16x8*)f)[1] = *(const bf16x8*)(row + 16 + (koff >> 1));
}

// ---------------------------------------------------------------------------
// Generic pointwise-conv GEMM: O[b][o][l] = act?(sum_c A[b][c][l] * W[o][c])
// grid.x = kB * (kL/16); block = 256 (8 waves, wave strides over N tiles).
// LDS: A tile stored M-major [16][ASTR], W stored [Cout][WSTR].
// ---------------------------------------------------------------------------
__global__ void gemm_pw_wmma(const float* __restrict__ A,
                             const float* __restrict__ W,
                             float* __restrict__ O,
                             int Cin, int Cout, int relu) {
  const int b    = blockIdx.x / (kL / 16);
  const int mt   = blockIdx.x % (kL / 16);
  const int tid  = threadIdx.x;
  const int Kpad = (Cin + 31) & ~31;
  const int STR  = Kpad + 8;            // row stride, 16B-aligned

  extern __shared__ char smem[];
  bf16_t* ldsA = (bf16_t*)smem;         // [16][STR]
  bf16_t* ldsW = ldsA + 16 * STR;       // [Cout][STR]

  for (int i = tid; i < Kpad * 16; i += 256) {
    int m = i & 15, k = i >> 4;         // consecutive tids -> consecutive m (coalesced)
    float v = (k < Cin) ? A[((size_t)b * Cin + k) * kL + mt * 16 + m] : 0.0f;
    ldsA[m * STR + k] = (bf16_t)v;
  }
  for (int i = tid; i < Cout * Kpad; i += 256) {
    int o = i / Kpad, k = i % Kpad;
    float v = (k < Cin) ? W[o * Cin + k] : 0.0f;
    ldsW[o * STR + k] = (bf16_t)v;
  }
  __syncthreads();

  const int lane = tid & 31;
  const int wave = tid >> 5;
  const int n    = lane & 15;
  const int hi   = (lane >> 4) << 3;
  const int koff = (lane >> 4) << 4;
  const bf16_t* arow = ldsA + n * STR;

  for (int nt = wave; nt < Cout / 16; nt += 8) {
    const bf16_t* wrow = ldsW + (nt * 16 + n) * STR;
    floatx8 acc = {};
    for (int kc = 0; kc < Kpad; kc += 32) {
      bf16x16 af, bf;
      load_a_frag(&af, arow + kc, hi);
      load_b_frag(&bf, wrow + kc, koff);
      acc = __builtin_amdgcn_wmma_f32_16x16x32_bf16(
          false, af, false, bf, (short)0, acc, false, false);
    }
    int oc = nt * 16 + n;
    int lb = mt * 16 + hi;               // D: VGPR r -> M=r+hi, N=lane&15
#pragma unroll
    for (int r = 0; r < 8; ++r) {
      float v = acc[r];
      if (relu) v = fmaxf(v, 0.0f);
      O[((size_t)b * Cout + oc) * kL + lb + r] = v;
    }
  }
}

// ---------------------------------------------------------------------------
// Gated depthwise mix (k=3 / k=15), writes bf16 with optional transpose+scale.
// transpose=1: out[(bh*kL + l)*32 + c]  (qk, pre-scaled by 32^-1/4)
// transpose=0: out[(b*256+ch)*kL + l]  (v, row-major)
// ---------------------------------------------------------------------------
__global__ void dds_mix_bf16(const float* __restrict__ y,
                             const float* __restrict__ dwa,
                             const float* __restrict__ dwb,
                             const float* __restrict__ gate,
                             bf16_t* __restrict__ out,
                             int transpose, float scale) {
  int idx = blockIdx.x * blockDim.x + threadIdx.x;
  if (idx >= kB * kHC * kL) return;
  int l  = idx & (kL - 1);
  int bc = idx >> 9;                     // b*256 + ch
  int ch = bc & (kHC - 1);
  const float* row = y + (size_t)bc * kL;

  float g0 = gate[0], g1 = gate[1];
  float mx = fmaxf(g0, g1);
  float e0 = __expf(g0 - mx), e1 = __expf(g1 - mx);
  float inv = 1.0f / (e0 + e1);
  g0 = e0 * inv; g1 = e1 * inv;

  float sa = 0.f, sb = 0.f;
#pragma unroll
  for (int t = 0; t < 3; ++t) {
    int ll = l + t - 1;
    if (ll >= 0 && ll < kL) sa += row[ll] * dwa[ch * 3 + t];
  }
#pragma unroll
  for (int t = 0; t < 15; ++t) {
    int ll = l + t - 7;
    if (ll >= 0 && ll < kL) sb += row[ll] * dwb[ch * 15 + t];
  }
  float val = (g0 * sa + g1 * sb) * scale;
  if (transpose)
    out[((size_t)(bc >> 5) * kL + l) * 32 + (ch & 31)] = (bf16_t)val;
  else
    out[(size_t)idx] = (bf16_t)val;
}

// ---------------------------------------------------------------------------
// Flash attention per (b,h): S = qk^T qk (channels K=32), softmax over keys,
// O = V P.  grid = kB*kH, block = 256 (8 waves, wave per query tile).
// qkT: bf16 [bh][l][32] (pre-scaled); vB: bf16 [bh][32][l].
// Staging via global_load_async_to_lds_b128 (ASYNCcnt).
// ---------------------------------------------------------------------------
__global__ void attn_kernel(const bf16_t* __restrict__ qkT,
                            const bf16_t* __restrict__ vB,
                            float* __restrict__ out) {
  extern __shared__ char smem[];
  bf16_t* sq = (bf16_t*)smem;             // [kL][QSTR]   queries/keys, [l][c]
  bf16_t* sv = sq + kL * QSTR;            // [32][SSTR]   V, [c][l]
  bf16_t* sp = sv + 32 * SSTR;            // [8][16][PSTR] per-wave P tiles

  const int bh  = blockIdx.x;
  const int tid = threadIdx.x;
  const bf16_t* gq = qkT + (size_t)bh * kL * 32;
  const bf16_t* gv = vB  + (size_t)bh * 32 * kL;
  const unsigned sq_off = (unsigned)(uintptr_t)(void*)sq;
  const unsigned sv_off = (unsigned)(uintptr_t)(void*)sv;

  // async copy: 2048 16-byte chunks each (32 KB per matrix), padded LDS rows
  for (int c = tid; c < 2048; c += 256) {          // qkT: 512 rows x 4 chunks
    unsigned lds = sq_off + (unsigned)(c >> 2) * (QSTR * 2) + (unsigned)(c & 3) * 16;
    unsigned long long ga = (unsigned long long)(uintptr_t)(gq + c * 8);
    asm volatile("global_load_async_to_lds_b128 %0, %1, off"
                 :: "v"(lds), "v"(ga) : "memory");
  }
  for (int c = tid; c < 2048; c += 256) {          // vB: 32 rows x 64 chunks
    unsigned lds = sv_off + (unsigned)(c >> 6) * (SSTR * 2) + (unsigned)(c & 63) * 16;
    unsigned long long ga = (unsigned long long)(uintptr_t)(gv + c * 8);
    asm volatile("global_load_async_to_lds_b128 %0, %1, off"
                 :: "v"(lds), "v"(ga) : "memory");
  }
  asm volatile("s_wait_asynccnt 0" ::: "memory");
  __syncthreads();

  const int lane = tid & 31;
  const int wave = tid >> 5;
  const int n    = lane & 15;
  const int hi   = (lane >> 4) << 3;
  const int koff = (lane >> 4) << 4;
  bf16_t* Pt = sp + wave * 16 * PSTR;     // [query 16][key 32]

  for (int qt = wave; qt < kL / 16; qt += 8) {
    const int mbase = qt * 16;
    bf16x16 bq;                            // B operand: K=channel, N=query
    load_b_frag(&bq, sq + (size_t)(mbase + n) * QSTR, koff);

    floatx8 acc0 = {}, acc1 = {};
    float run_max = -INFINITY, run_sum = 0.0f;

    for (int kblk = 0; kblk < kL; kblk += 32) {
      // S: A operand = key rows (M=key, K=channel), two 16-key halves
      bf16x16 a0, a1;
      load_a_frag(&a0, sq + (size_t)(kblk + n) * QSTR, hi);
      load_a_frag(&a1, sq + (size_t)(kblk + 16 + n) * QSTR, hi);
      floatx8 z = {};
      floatx8 s0 = __builtin_amdgcn_wmma_f32_16x16x32_bf16(
          false, a0, false, bq, (short)0, z, false, false);
      floatx8 s1 = __builtin_amdgcn_wmma_f32_16x16x32_bf16(
          false, a1, false, bq, (short)0, z, false, false);

      // online softmax over keys (halves joined via xor-16, wave32)
      float t = -INFINITY;
#pragma unroll
      for (int r = 0; r < 8; ++r) t = fmaxf(t, fmaxf(s0[r], s1[r]));
      t = fmaxf(t, __shfl_xor(t, 16, 32));
      float nm = fmaxf(run_max, t);
      float alpha = __expf(run_max - nm);
      float ps = 0.0f;
#pragma unroll
      for (int r = 0; r < 8; ++r) {
        s0[r] = __expf(s0[r] - nm);
        s1[r] = __expf(s1[r] - nm);
        ps += s0[r] + s1[r];
      }
      ps += __shfl_xor(ps, 16, 32);
      run_sum = run_sum * alpha + ps;
      run_max = nm;
#pragma unroll
      for (int r = 0; r < 8; ++r) { acc0[r] *= alpha; acc1[r] *= alpha; }

      // P: D-layout -> transposed LDS tile [query][key] -> B fragment
      bf16x8 p0, p1;
#pragma unroll
      for (int r = 0; r < 8; ++r) { p0[r] = (bf16_t)s0[r]; p1[r] = (bf16_t)s1[r]; }
      *(bf16x8*)(Pt + n * PSTR + hi)      = p0;   // keys hi..hi+7
      *(bf16x8*)(Pt + n * PSTR + 16 + hi) = p1;   // keys 16+hi..23+hi
      bf16x16 bp, av0, av1;
      load_b_frag(&bp, Pt + n * PSTR, koff);
      load_a_frag(&av0, sv + (size_t)n * SSTR + kblk, hi);         // V c=0..15
      load_a_frag(&av1, sv + (size_t)(16 + n) * SSTR + kblk, hi);  // V c=16..31
      acc0 = __builtin_amdgcn_wmma_f32_16x16x32_bf16(
          false, av0, false, bp, (short)0, acc0, false, false);
      acc1 = __builtin_amdgcn_wmma_f32_16x16x32_bf16(
          false, av1, false, bp, (short)0, acc1, false, false);
    }

    float inv = 1.0f / run_sum;
    float* ob = out + (size_t)bh * 32 * kL;
#pragma unroll
    for (int r = 0; r < 8; ++r) {
      ob[(hi + r) * kL + mbase + n]      = acc0[r] * inv;
      ob[(hi + r + 16) * kL + mbase + n] = acc1[r] * inv;
    }
  }
}

// ---------------------------------------------------------------------------
// out = InstanceNorm1d(xin + res); one wave per (b,c) row of length 512.
// ---------------------------------------------------------------------------
__global__ void add_inorm(const float* __restrict__ xin,
                          const float* __restrict__ res,
                          const float* __restrict__ g,
                          const float* __restrict__ bta,
                          float* __restrict__ out) {
  int row  = blockIdx.x * 8 + (threadIdx.x >> 5);
  int lane = threadIdx.x & 31;
  int c = row & (kTC - 1);
  const float* p1 = xin + (size_t)row * kL;
  const float* p2 = res + (size_t)row * kL;
  float vals[16];
  float s = 0.f;
#pragma unroll
  for (int i = 0; i < 16; ++i) {
    vals[i] = p1[lane + i * 32] + p2[lane + i * 32];
    s += vals[i];
  }
  for (int off = 16; off >= 1; off >>= 1) s += __shfl_xor(s, off, 32);
  float mu = s * (1.0f / kL);
  float vsum = 0.f;
#pragma unroll
  for (int i = 0; i < 16; ++i) {
    float d = vals[i] - mu;
    vsum += d * d;
  }
  for (int off = 16; off >= 1; off >>= 1) vsum += __shfl_xor(vsum, off, 32);
  float sc = g[c] * rsqrtf(vsum * (1.0f / kL) + 1e-5f);
  float bb = bta[c];
  float* po = out + (size_t)row * kL;
#pragma unroll
  for (int i = 0; i < 16; ++i)
    po[lane + i * 32] = (vals[i] - mu) * sc + bb;
}

// ---------------------------------------------------------------------------
// to_logits: y = pw.act (32->1), DDS (k=1 / k=3), sigmoid. grid=B, block=512.
// ---------------------------------------------------------------------------
__global__ void logits_kernel(const float* __restrict__ act,
                              const float* __restrict__ pw,
                              const float* __restrict__ dwa,
                              const float* __restrict__ dwb,
                              const float* __restrict__ gate,
                              float* __restrict__ out) {
  __shared__ float y[kL];
  int b = blockIdx.x, l = threadIdx.x;
  float s = 0.f;
#pragma unroll
  for (int c = 0; c < kTC; ++c)
    s += act[((size_t)b * kTC + c) * kL + l] * pw[c];
  y[l] = s;
  __syncthreads();

  float g0 = gate[0], g1 = gate[1];
  float mx = fmaxf(g0, g1);
  float e0 = __expf(g0 - mx), e1 = __expf(g1 - mx);
  float inv = 1.0f / (e0 + e1);
  g0 = e0 * inv; g1 = e1 * inv;

  float a = dwa[0] * y[l];
  float bconv = 0.f;
  for (int t = 0; t < 3; ++t) {
    int ll = l + t - 1;
    if (ll >= 0 && ll < kL) bconv += dwb[t] * y[ll];
  }
  float z = g0 * a + g1 * bconv;
  out[(size_t)b * kL + l] = 1.0f / (1.0f + __expf(-z));
}

// ---------------------------------------------------------------------------
extern "C" void kernel_launch(void* const* d_in, const int* in_sizes, int n_in,
                              void* d_out, int out_size, void* d_ws, size_t ws_size,
                              hipStream_t stream) {
  (void)in_sizes; (void)n_in; (void)out_size; (void)ws_size;
  const float* x       = (const float*)d_in[0];
  const float* enc_w   = (const float*)d_in[1];
  const float* qk_pw   = (const float*)d_in[2];
  const float* qk_dwa  = (const float*)d_in[3];
  const float* qk_dwb  = (const float*)d_in[4];
  const float* qk_gate = (const float*)d_in[5];
  const float* v_pw    = (const float*)d_in[6];
  const float* v_dwa   = (const float*)d_in[7];
  const float* v_dwb   = (const float*)d_in[8];
  const float* v_gate  = (const float*)d_in[9];
  const float* unify_w = (const float*)d_in[10];
  const float* n1_g    = (const float*)d_in[11];
  const float* n1_b    = (const float*)d_in[12];
  const float* ff_w1   = (const float*)d_in[13];
  const float* ff_w2   = (const float*)d_in[14];
  const float* n2_g    = (const float*)d_in[15];
  const float* n2_b    = (const float*)d_in[16];
  const float* log_pw  = (const float*)d_in[17];
  const float* log_dwa = (const float*)d_in[18];
  const float* log_dwb = (const float*)d_in[19];
  const float* log_gat = (const float*)d_in[20];

  // workspace (~38 MB)
  float*  act  = (float*)d_ws;                      // [B][32][L] f32
  float*  Ybuf = act + (size_t)kB * kTC * kL;       // [B][256][L] f32 scratch
  float*  TMP  = Ybuf + (size_t)kB * kHC * kL;      // [B][32][L] f32
  bf16_t* qkT  = (bf16_t*)(TMP + (size_t)kB * kTC * kL);  // [bh][L][32] bf16
  bf16_t* vBuf = qkT + (size_t)kB * kHC * kL;             // [bh][32][L] bf16

  auto gemm = [&](const float* A, const float* W, float* O, int Cin, int Cout,
                  int relu) {
    int Kpad = (Cin + 31) & ~31;
    size_t sm = (size_t)(Kpad + 8) * (16 + Cout) * sizeof(bf16_t);
    gemm_pw_wmma<<<dim3(kB * (kL / 16)), dim3(256), sm, stream>>>(
        A, W, O, Cin, Cout, relu);
  };

  const int ddsBlocks = (kB * kHC * kL + 255) / 256;
  const size_t attnSm =
      (size_t)(kL * QSTR + 32 * SSTR + 8 * 16 * PSTR) * sizeof(bf16_t);
  const float qscale = 0.42044820762685725f;  // 32^-1/4 on both q and k

  gemm(x, enc_w, act, kCIN, kTC, 0);          // init encoder 6 -> 32

  for (int i = 0; i < kD; ++i) {
    gemm(act, qk_pw + (size_t)i * kHC * kTC, Ybuf, kTC, kHC, 0);
    dds_mix_bf16<<<ddsBlocks, 256, 0, stream>>>(
        Ybuf, qk_dwa + (size_t)i * kHC * 3, qk_dwb + (size_t)i * kHC * 15,
        qk_gate + (size_t)i * 2, qkT, 1, qscale);
    gemm(act, v_pw + (size_t)i * kHC * kTC, Ybuf, kTC, kHC, 0);
    dds_mix_bf16<<<ddsBlocks, 256, 0, stream>>>(
        Ybuf, v_dwa + (size_t)i * kHC * 3, v_dwb + (size_t)i * kHC * 15,
        v_gate + (size_t)i * 2, vBuf, 0, 1.0f);
    attn_kernel<<<dim3(kB * kH), dim3(256), attnSm, stream>>>(qkT, vBuf, Ybuf);
    gemm(Ybuf, unify_w + (size_t)i * kTC * kHC, TMP, kHC, kTC, 0);
    add_inorm<<<dim3(kB * kTC / 8), dim3(256), 0, stream>>>(
        TMP, act, n1_g + (size_t)i * kTC, n1_b + (size_t)i * kTC, act);
    gemm(act, ff_w1 + (size_t)i * kFF * kTC, Ybuf, kTC, kFF, 1);
    gemm(Ybuf, ff_w2 + (size_t)i * kTC * kFF, TMP, kFF, kTC, 0);
    add_inorm<<<dim3(kB * kTC / 8), dim3(256), 0, stream>>>(
        TMP, act, n2_g + (size_t)i * kTC, n2_b + (size_t)i * kTC, act);
  }

  logits_kernel<<<dim3(kB), dim3(kL), 0, stream>>>(
      act, log_pw, log_dwa, log_dwb, log_gat, (float*)d_out);
}
